// TCRLSTM_48249662604051
// MI455X (gfx1250) — matrix operand order repaired and checked by reading, not compiled, not run on hardware
//
#include <hip/hip_runtime.h>
#include <stdint.h>

#define BB 2048
#define LL 35
#define EE 128
#define HH 1024
#define VV 21
#define TT 36          // L+1 steps
#define KK 1152        // E+H
#define KC (KK/32)     // 36 K-chunks of 32
#define NT16 (HH/16)   // 64 column tiles of 16

typedef __attribute__((ext_vector_type(16))) __bf16 v16bf;
typedef __attribute__((ext_vector_type(8)))  float  v8f;
typedef __attribute__((ext_vector_type(4)))  unsigned int u32x4;

union FragU { u32x4 u[2]; v16bf v; };

__device__ __forceinline__ unsigned short f2bf(float x) {
  unsigned u = __float_as_uint(x);
  unsigned r = (u + 0x7fffu + ((u >> 16) & 1u)) >> 16;   // RNE
  return (unsigned short)r;
}
__device__ __forceinline__ float bf2f(unsigned short h) {
  return __uint_as_float(((unsigned)h) << 16);
}
__device__ __forceinline__ float sigmoidf_(float x) { return 1.0f / (1.0f + __expf(-x)); }

__device__ __forceinline__ v8f zero8() {
  v8f z;
  #pragma unroll
  for (int i = 0; i < 8; ++i) z[i] = 0.0f;
  return z;
}

__device__ __forceinline__ v8f wmma_bf16(v16bf a, v16bf b, v8f c) {
  // D = A(16x32 bf16) * B(32x16 bf16) + C(16x16 f32)
  return __builtin_amdgcn_wmma_f32_16x16x32_bf16(false, a, false, b, (short)0, c, false, false);
}

// A fragment: lane holds K [kb..kb+15], kb=(lane>=16)?16:0 (matches packed B)
__device__ __forceinline__ void load_a(FragU& f, const unsigned short* xrow,
                                       const unsigned short* hrow, int kc, int aSub) {
  const int kbase = kc * 32 + aSub;
  const unsigned short* ap = (kc < 4) ? (xrow + kbase) : (hrow + (kbase - EE));
  f.u[0] = *(const u32x4*)(ap);
  f.u[1] = *(const u32x4*)(ap + 8);
}
// B fragment: 32 contiguous bytes per lane (pre-packed)
__device__ __forceinline__ void load_b(FragU& f, const unsigned short* p) {
  f.u[0] = *(const u32x4*)(p);
  f.u[1] = *(const u32x4*)(p + 8);
}

// ---------------- init: c state, h0 ----------------
__global__ void __launch_bounds__(256)
init_kernel(float* __restrict__ Cst, unsigned short* __restrict__ Hhist0) {
  size_t i = (size_t)blockIdx.x * 256 + threadIdx.x;
  if (i < (size_t)BB * HH) { Cst[i] = 0.0f; Hhist0[i] = 0; }
}

// ---------------- embedding: Xbf[t,b,e] = (t==0)?0:emb[tok[b][t-1]][e] (bf16) ----------------
__global__ void __launch_bounds__(256)
embed_kernel(const int* __restrict__ tokens, const float* __restrict__ emb,
             unsigned short* __restrict__ Xbf) {
  size_t idx = (size_t)blockIdx.x * 256 + threadIdx.x;
  if (idx >= (size_t)TT * BB * EE) return;
  int e = (int)(idx & (EE - 1));
  size_t rb = idx >> 7;                  // /EE
  int t = (int)(rb / BB);
  int b = (int)(rb - (size_t)t * BB);
  float v = 0.0f;
  if (t > 0) {
    int tok = tokens[b * LL + (t - 1)];
    v = emb[(size_t)tok * EE + e];
  }
  Xbf[idx] = f2bf(v);
}

// ------------- pack Wi/Wo/Wf/Wg/W1 (fp32 [K,1024]) into B-fragment order (bf16) -------------
// Fragment tile (kc,nT): 32 lanes x 16 bf16. lane: col = nT*16+(lane&15), K = kc*32+(lane>>4)*16 + j
__global__ void __launch_bounds__(256)
pack_kernel(const float* __restrict__ Wi, const float* __restrict__ Wo,
            const float* __restrict__ Wf_, const float* __restrict__ Wg,
            const float* __restrict__ W1, unsigned short* __restrict__ Wpack) {
  int u = blockIdx.x * 256 + threadIdx.x;
  const int total = 5 * KC * NT16 * 32;
  if (u >= total) return;
  int lane = u & 31; u >>= 5;
  int nT = u % NT16; u /= NT16;
  int kc = u % KC;  int g = u / KC;
  const float* src = (g == 0) ? Wi : (g == 1) ? Wo : (g == 2) ? Wf_ : (g == 3) ? Wg : W1;
  int col = nT * 16 + (lane & 15);
  int kb  = kc * 32 + ((lane >> 4) * 16);
  unsigned short* dst = Wpack + ((size_t)((g * KC + kc) * NT16 + nT)) * 512 + lane * 16;
  #pragma unroll
  for (int j = 0; j < 16; ++j) dst[j] = f2bf(src[(size_t)(kb + j) * HH + col]);
}

// ---------------- one LSTM time step: gates GEMM (WMMA) + cell update ----------------
// grid (BB/256, NT16); block 256 (8 waves).
// Wave w: rows [bx*256 + w*32, +32) (2 A-frags), cols [nT*16,+16), 4 gates -> 8 WMMA / k-chunk.
// Software-pipelined: ping-pong fragment sets so loads for kc+1 are in flight during WMMAs for kc.
__global__ void __launch_bounds__(256)
lstm_step_kernel(const unsigned short* __restrict__ Xbf,   // [T,B,E] bf16
                 unsigned short* __restrict__ Hhist,       // [T+1,B,H] bf16
                 float* __restrict__ Cst,                  // [B,H] f32
                 const unsigned short* __restrict__ Wpack, // gates i,o,f,g packed
                 const float* __restrict__ bi, const float* __restrict__ bo,
                 const float* __restrict__ bfv, const float* __restrict__ bg,
                 int t) {
  const int lane = threadIdx.x & 31;
  const int wid  = threadIdx.x >> 5;
  const int m0   = blockIdx.x * 256 + wid * 32;
  const int nT   = blockIdx.y;

  const int rowA0 = m0 + (lane & 15);
  const int rowA1 = rowA0 + 16;
  const int aSub  = (lane >> 4) * 16;

  const unsigned short* xrow0 = Xbf   + ((size_t)t * BB + rowA0) * EE;
  const unsigned short* hrow0 = Hhist + ((size_t)t * BB + rowA0) * HH;
  const unsigned short* xrow1 = Xbf   + ((size_t)t * BB + rowA1) * EE;
  const unsigned short* hrow1 = Hhist + ((size_t)t * BB + rowA1) * HH;

  const size_t gstride = (size_t)KC * NT16 * 512;
  const unsigned short* bcol = Wpack + (size_t)nT * 512 + lane * 16;

  v8f acc[4][2];
  #pragma unroll
  for (int g = 0; g < 4; ++g) { acc[g][0] = zero8(); acc[g][1] = zero8(); }

  FragU A0[2], B0[4], A1[2], B1[4];

  // prologue: set0 <- kc=0
  load_a(A0[0], xrow0, hrow0, 0, aSub);
  load_a(A0[1], xrow1, hrow1, 0, aSub);
  #pragma unroll
  for (int g = 0; g < 4; ++g) load_b(B0[g], bcol + g * gstride);

  for (int kc = 0; kc < KC - 2; kc += 2) {
    // prefetch set1 <- kc+1
    load_a(A1[0], xrow0, hrow0, kc + 1, aSub);
    load_a(A1[1], xrow1, hrow1, kc + 1, aSub);
    #pragma unroll
    for (int g = 0; g < 4; ++g) load_b(B1[g], bcol + (size_t)(kc + 1) * (NT16 * 512) + g * gstride);
    // compute set0 (kc)
    #pragma unroll
    for (int g = 0; g < 4; ++g) {
      acc[g][0] = wmma_bf16(A0[0].v, B0[g].v, acc[g][0]);
      acc[g][1] = wmma_bf16(A0[1].v, B0[g].v, acc[g][1]);
    }
    // prefetch set0 <- kc+2
    load_a(A0[0], xrow0, hrow0, kc + 2, aSub);
    load_a(A0[1], xrow1, hrow1, kc + 2, aSub);
    #pragma unroll
    for (int g = 0; g < 4; ++g) load_b(B0[g], bcol + (size_t)(kc + 2) * (NT16 * 512) + g * gstride);
    // compute set1 (kc+1)
    #pragma unroll
    for (int g = 0; g < 4; ++g) {
      acc[g][0] = wmma_bf16(A1[0].v, B1[g].v, acc[g][0]);
      acc[g][1] = wmma_bf16(A1[1].v, B1[g].v, acc[g][1]);
    }
  }
  // tail: set0 holds KC-2; load set1 <- KC-1, then compute both
  load_a(A1[0], xrow0, hrow0, KC - 1, aSub);
  load_a(A1[1], xrow1, hrow1, KC - 1, aSub);
  #pragma unroll
  for (int g = 0; g < 4; ++g) load_b(B1[g], bcol + (size_t)(KC - 1) * (NT16 * 512) + g * gstride);
  #pragma unroll
  for (int g = 0; g < 4; ++g) {
    acc[g][0] = wmma_bf16(A0[0].v, B0[g].v, acc[g][0]);
    acc[g][1] = wmma_bf16(A0[1].v, B0[g].v, acc[g][1]);
    acc[g][0] = wmma_bf16(A1[0].v, B1[g].v, acc[g][0]);
    acc[g][1] = wmma_bf16(A1[1].v, B1[g].v, acc[g][1]);
  }

  // epilogue: gate math in fp32, cell update, h history in bf16
  const int n = nT * 16 + (lane & 15);
  const float bin = bi[n], bon = bo[n], bfn = bfv[n], bgn = bg[n];
  const int half = (lane < 16) ? 0 : 8;

  #pragma unroll
  for (int mi = 0; mi < 2; ++mi) {
    #pragma unroll
    for (int r = 0; r < 8; ++r) {
      const int m = m0 + mi * 16 + r + half;
      float iv = sigmoidf_(acc[0][mi][r] + bin);
      float ov = sigmoidf_(acc[1][mi][r] + bon);
      float fv = sigmoidf_(acc[2][mi][r] + bfn);
      float gv = tanhf(acc[3][mi][r] + bgn);
      size_t cidx = (size_t)m * HH + n;
      float cn = fv * Cst[cidx] + iv * gv;
      Cst[cidx] = cn;
      float hn = ov * tanhf(cn);
      Hhist[((size_t)(t + 1) * BB + m) * HH + n] = f2bf(hn);
    }
  }
}

// ---------------- MLP layer 1: Y = relu(xh @ W1 + b1) over all T*B rows ----------------
// grid (T*B/256, H/64); block 256. Wave w: rows [bx*256+w*32,+32), cols [by*64,+64) (4 n-frags).
__global__ void __launch_bounds__(256)
mlp1_kernel(const unsigned short* __restrict__ Xbf,   // [T*B, E]
            const unsigned short* __restrict__ Hhist, // [(T+1)*B, H], slots 0..T-1
            const unsigned short* __restrict__ W1p,
            const float* __restrict__ b1,
            unsigned short* __restrict__ Y) {
  const int lane = threadIdx.x & 31;
  const int wid  = threadIdx.x >> 5;
  const int m0   = blockIdx.x * 256 + wid * 32;
  const int nT4  = blockIdx.y;

  const int rowA0 = m0 + (lane & 15);
  const int rowA1 = rowA0 + 16;
  const int aSub  = (lane >> 4) * 16;

  const unsigned short* xrow0 = Xbf   + (size_t)rowA0 * EE;
  const unsigned short* hrow0 = Hhist + (size_t)rowA0 * HH;
  const unsigned short* xrow1 = Xbf   + (size_t)rowA1 * EE;
  const unsigned short* hrow1 = Hhist + (size_t)rowA1 * HH;

  const unsigned short* bcol = W1p + (size_t)(nT4 * 4) * 512 + lane * 16;

  v8f acc[4][2];
  #pragma unroll
  for (int s = 0; s < 4; ++s) { acc[s][0] = zero8(); acc[s][1] = zero8(); }

  FragU A0[2], B0[4], A1[2], B1[4];

  load_a(A0[0], xrow0, hrow0, 0, aSub);
  load_a(A0[1], xrow1, hrow1, 0, aSub);
  #pragma unroll
  for (int s = 0; s < 4; ++s) load_b(B0[s], bcol + s * 512);

  for (int kc = 0; kc < KC - 2; kc += 2) {
    load_a(A1[0], xrow0, hrow0, kc + 1, aSub);
    load_a(A1[1], xrow1, hrow1, kc + 1, aSub);
    #pragma unroll
    for (int s = 0; s < 4; ++s) load_b(B1[s], bcol + (size_t)(kc + 1) * (NT16 * 512) + s * 512);
    #pragma unroll
    for (int s = 0; s < 4; ++s) {
      acc[s][0] = wmma_bf16(A0[0].v, B0[s].v, acc[s][0]);
      acc[s][1] = wmma_bf16(A0[1].v, B0[s].v, acc[s][1]);
    }
    load_a(A0[0], xrow0, hrow0, kc + 2, aSub);
    load_a(A0[1], xrow1, hrow1, kc + 2, aSub);
    #pragma unroll
    for (int s = 0; s < 4; ++s) load_b(B0[s], bcol + (size_t)(kc + 2) * (NT16 * 512) + s * 512);
    #pragma unroll
    for (int s = 0; s < 4; ++s) {
      acc[s][0] = wmma_bf16(A1[0].v, B1[s].v, acc[s][0]);
      acc[s][1] = wmma_bf16(A1[1].v, B1[s].v, acc[s][1]);
    }
  }
  load_a(A1[0], xrow0, hrow0, KC - 1, aSub);
  load_a(A1[1], xrow1, hrow1, KC - 1, aSub);
  #pragma unroll
  for (int s = 0; s < 4; ++s) load_b(B1[s], bcol + (size_t)(KC - 1) * (NT16 * 512) + s * 512);
  #pragma unroll
  for (int s = 0; s < 4; ++s) {
    acc[s][0] = wmma_bf16(A0[0].v, B0[s].v, acc[s][0]);
    acc[s][1] = wmma_bf16(A0[1].v, B0[s].v, acc[s][1]);
    acc[s][0] = wmma_bf16(A1[0].v, B1[s].v, acc[s][0]);
    acc[s][1] = wmma_bf16(A1[1].v, B1[s].v, acc[s][1]);
  }

  const int half = (lane < 16) ? 0 : 8;
  #pragma unroll
  for (int s = 0; s < 4; ++s) {
    const int n = nT4 * 64 + s * 16 + (lane & 15);
    const float bn = b1[n];
    #pragma unroll
    for (int mi = 0; mi < 2; ++mi) {
      #pragma unroll
      for (int r = 0; r < 8; ++r) {
        const int m = m0 + mi * 16 + r + half;
        float y = acc[s][mi][r] + bn;
        y = (y > 0.0f) ? y : 0.0f;
        Y[(size_t)m * HH + n] = f2bf(y);
      }
    }
  }
}

// ---------------- classifier GEMM (N=21) + log-softmax + NLL + argmax, one wave per row ----------------
__global__ void __launch_bounds__(256)
loss_kernel(const unsigned short* __restrict__ Y, const float* __restrict__ W2,
            const float* __restrict__ b2, const int* __restrict__ tokens,
            const int* __restrict__ lengths,
            float* __restrict__ rowLoss, float* __restrict__ rowCorr) {
  __shared__ unsigned short sW2[HH * VV];   // 43 KB bf16 copy of W2 (CDNA5 LDS: 320KB/WGP)
  for (int i = threadIdx.x; i < HH * VV; i += 256) sW2[i] = f2bf(W2[i]);
  __syncthreads();

  const int lane = threadIdx.x & 31;
  const int wid  = threadIdx.x >> 5;
  const int r = blockIdx.x * 8 + wid;       // row in [0, T*B)
  const int t = r / BB;
  const int b = r - t * BB;
  const int len = lengths[b];

  float nll = 0.0f, corr = 0.0f;
  if (t <= len) {
    union { u32x4 u[4]; unsigned short us[32]; } yb;
    const unsigned short* yp = Y + (size_t)r * HH + lane * 32;
    yb.u[0] = *(const u32x4*)(yp);      yb.u[1] = *(const u32x4*)(yp + 8);
    yb.u[2] = *(const u32x4*)(yp + 16); yb.u[3] = *(const u32x4*)(yp + 24);
    float yv[32];
    #pragma unroll
    for (int j = 0; j < 32; ++j) yv[j] = bf2f(yb.us[j]);

    float logit[VV];
    #pragma unroll
    for (int v = 0; v < VV; ++v) {
      float s = 0.0f;
      #pragma unroll
      for (int j = 0; j < 32; ++j)
        s = fmaf(yv[j], bf2f(sW2[(lane * 32 + j) * VV + v]), s);
      #pragma unroll
      for (int off = 16; off > 0; off >>= 1) s += __shfl_xor(s, off, 32);
      logit[v] = s + b2[v];
    }
    float mx = logit[0];
    #pragma unroll
    for (int v = 1; v < VV; ++v) mx = fmaxf(mx, logit[v]);
    float se = 0.0f;
    #pragma unroll
    for (int v = 0; v < VV; ++v) se += __expf(logit[v] - mx);
    float lse = logf(se) + mx;

    int tg = (t < LL) ? tokens[b * LL + t] : 0;
    float ltg = 0.0f;
    #pragma unroll
    for (int v = 0; v < VV; ++v) if (v == tg) ltg = logit[v];
    nll = lse - ltg;

    int pred = 0; float bv = logit[0];
    #pragma unroll
    for (int v = 1; v < VV; ++v) if (logit[v] > bv) { bv = logit[v]; pred = v; }
    corr = (pred == tg) ? 1.0f : 0.0f;
  }
  if (lane == 0) { rowLoss[r] = nll; rowCorr[r] = corr; }
}

// ---------------- deterministic final reduction ----------------
__global__ void __launch_bounds__(256)
reduce_kernel(const float* __restrict__ rowLoss, const float* __restrict__ rowCorr,
              const int* __restrict__ lengths, float* __restrict__ out) {
  __shared__ float sl[256], sc[256], sn[256];
  float a = 0.0f, c = 0.0f, n = 0.0f;
  for (int i = threadIdx.x; i < TT * BB; i += 256) { a += rowLoss[i]; c += rowCorr[i]; }
  for (int i = threadIdx.x; i < BB; i += 256) n += (float)(lengths[i] + 1);
  sl[threadIdx.x] = a; sc[threadIdx.x] = c; sn[threadIdx.x] = n;
  __syncthreads();
  for (int s = 128; s > 0; s >>= 1) {
    if (threadIdx.x < s) {
      sl[threadIdx.x] += sl[threadIdx.x + s];
      sc[threadIdx.x] += sc[threadIdx.x + s];
      sn[threadIdx.x] += sn[threadIdx.x + s];
    }
    __syncthreads();
  }
  if (threadIdx.x == 0) {
    out[0] = sl[0] / (float)BB;
    out[1] = sc[0] / sn[0];
  }
}

extern "C" void kernel_launch(void* const* d_in, const int* in_sizes, int n_in,
                              void* d_out, int out_size, void* d_ws, size_t ws_size,
                              hipStream_t stream) {
  const int*   tokens  = (const int*)  d_in[0];
  const int*   lengths = (const int*)  d_in[1];
  const float* emb     = (const float*)d_in[2];
  const float* Wi = (const float*)d_in[3];  const float* bi = (const float*)d_in[4];
  const float* Wo = (const float*)d_in[5];  const float* bo = (const float*)d_in[6];
  const float* Wf = (const float*)d_in[7];  const float* bf = (const float*)d_in[8];
  const float* Wg = (const float*)d_in[9];  const float* bg = (const float*)d_in[10];
  const float* W1 = (const float*)d_in[11]; const float* b1 = (const float*)d_in[12];
  const float* W2 = (const float*)d_in[13]; const float* b2 = (const float*)d_in[14];
  float* out = (float*)d_out;
  (void)in_sizes; (void)n_in; (void)out_size; (void)ws_size;

  char* ws = (char*)d_ws;
  size_t off = 0;
  auto take = [&](size_t bytes) -> char* {
    char* p = ws + off;
    off = (off + bytes + 255) & ~(size_t)255;
    return p;
  };
  unsigned short* Xbf   = (unsigned short*)take((size_t)TT * BB * EE * 2);
  unsigned short* Hhist = (unsigned short*)take((size_t)(TT + 1) * BB * HH * 2);
  float*          Cst   = (float*)        take((size_t)BB * HH * 4);
  unsigned short* Wpack = (unsigned short*)take((size_t)5 * KC * NT16 * 512 * 2);
  unsigned short* Ybuf  = (unsigned short*)take((size_t)TT * BB * HH * 2);
  float*          rowLoss = (float*)take((size_t)TT * BB * 4);
  float*          rowCorr = (float*)take((size_t)TT * BB * 4);

  init_kernel<<<(BB * HH + 255) / 256, 256, 0, stream>>>(Cst, Hhist);
  embed_kernel<<<(TT * BB * EE + 255) / 256, 256, 0, stream>>>(tokens, emb, Xbf);
  pack_kernel<<<(5 * KC * NT16 * 32 + 255) / 256, 256, 0, stream>>>(Wi, Wo, Wf, Wg, W1, Wpack);

  for (int t = 0; t < TT; ++t) {
    lstm_step_kernel<<<dim3(BB / 256, NT16), 256, 0, stream>>>(
        Xbf, Hhist, Cst, Wpack, bi, bo, bf, bg, t);
  }

  const unsigned short* W1p = Wpack + (size_t)4 * KC * NT16 * 512;
  mlp1_kernel<<<dim3(TT * BB / 256, HH / 64), 256, 0, stream>>>(Xbf, Hhist, W1p, b1, Ybuf);
  loss_kernel<<<TT * BB / 8, 256, 0, stream>>>(Ybuf, W2, b2, tokens, lengths, rowLoss, rowCorr);
  reduce_kernel<<<1, 256, 0, stream>>>(rowLoss, rowCorr, lengths, out);
}
